// VARNN_17205638988183
// MI455X (gfx1250) — matrix operand order repaired and checked by reading, not compiled, not run on hardware
//
#include <hip/hip_runtime.h>
#include <hip/hip_bf16.h>

typedef __attribute__((ext_vector_type(16))) __bf16 v16bf;
typedef __attribute__((ext_vector_type(8)))  __bf16 v8bf;
typedef __attribute__((ext_vector_type(8)))  float  v8f;
typedef __attribute__((ext_vector_type(4)))  unsigned v4u;
typedef __attribute__((ext_vector_type(8)))  int      v8i_;
typedef __attribute__((ext_vector_type(4)))  int      v4i_;

#define NB   256
#define TT   300
#define SS   150
#define HH   100
#define G4   400          // 4*H
#define NCLS 60
#define NT   (NB * TT)    // 76800

static __device__ __forceinline__ v16bf cat16(v8bf x, v8bf y) {
    return __builtin_shufflevector(x, y, 0,1,2,3,4,5,6,7,8,9,10,11,12,13,14,15);
}

// ---------------------------------------------------------------------------
// Activations: f32 [M][K] -> bf16 [M][Kpad] zero-padded (coalesced)
// ---------------------------------------------------------------------------
__global__ void k_convert_act(const float* __restrict__ src, __bf16* __restrict__ dst,
                              int M, int K, int Kpad) {
    int i = blockIdx.x * blockDim.x + threadIdx.x;
    if (i >= M * Kpad) return;
    int m = i / Kpad, k = i % Kpad;
    dst[i] = (__bf16)((k < K) ? src[(size_t)m * K + k] : 0.0f);
}

// ---------------------------------------------------------------------------
// Weights: W [G4][K] f32 -> fragment-swizzled bf16:
//   Ws[(((kc32*25)+nt)*32 + lane)*16 + e],  k = kc32*32 + (lane>>4)*16 + e,
//   n = nt*16 + (lane&15).  B-fragment load = 2 contiguous v8bf per lane.
// ---------------------------------------------------------------------------
__global__ void k_convert_w_swz(const float* __restrict__ W, __bf16* __restrict__ Ws,
                                int K, int KC32) {
    int i = blockIdx.x * blockDim.x + threadIdx.x;
    int total = KC32 * 25 * 512;
    if (i >= total) return;
    int e    = i & 15;
    int lane = (i >> 4) & 31;
    int nt   = (i >> 9) % 25;
    int kc32 = i / (25 * 512);
    int k = kc32 * 32 + (lane >> 4) * 16 + e;
    int g = nt * 16 + (lane & 15);
    float v = (k < K) ? W[g * K + k] : 0.0f;
    Ws[i] = (__bf16)v;
}

__global__ void k_bias_add(const float* __restrict__ a, const float* __restrict__ b,
                           float* __restrict__ out, int n) {
    int i = blockIdx.x * blockDim.x + threadIdx.x;
    if (i < n) out[i] = a[i] + b[i];
}

// ---------------------------------------------------------------------------
// Input projection: C[M][400] = Abf[M][Kpad] * W + bias. One wave = one 16x16
// tile; fully unrolled k-loop; all fragment feeds are wide unconditional loads.
// ---------------------------------------------------------------------------
template<int KC32>
__global__ __launch_bounds__(32) void k_gemm_bf16(
    const __bf16* __restrict__ Abf, const __bf16* __restrict__ Bswz,
    const float* __restrict__ bias, float* __restrict__ C)
{
    const int mBase = blockIdx.x * 16;
    const int nT    = blockIdx.y;                 // 0..24
    const int lane  = threadIdx.x;
    const int hi = lane >> 4, lo = lane & 15;
    constexpr int Kpad = KC32 * 32;

    const __bf16* arow  = Abf + (size_t)(mBase + lo) * Kpad + hi * 16;
    const __bf16* bbase = Bswz + ((size_t)nT * 32 + lane) * 16;

    v8f acc = {};
    #pragma unroll
    for (int c = 0; c < KC32; ++c) {
        const v8bf* ap = (const v8bf*)(arow + c * 32);
        v16bf a = cat16(ap[0], ap[1]);            // K = c*32 + hi*16 + e
        const v8bf* bp = (const v8bf*)(bbase + (size_t)c * 25 * 512);
        v16bf b = cat16(bp[0], bp[1]);            // K = c*32 + hi*16 + e
        acc = __builtin_amdgcn_wmma_f32_16x16x32_bf16(false, a, false, b,
                                                      (short)0, acc, false, false);
    }
    float bs = bias[nT * 16 + lo];
    #pragma unroll
    for (int v = 0; v < 8; ++v)
        C[(size_t)(mBase + v + hi * 8) * G4 + nT * 16 + lo] = acc[v] + bs;
}

// ---------------------------------------------------------------------------
// LSTM recurrence. Grid = 16 WGs (16 batch rows each), block = 160 (5 waves).
// Whh (fragment-swizzled bf16, 100KB) DMA'd into LDS by the Tensor Data Mover;
// h lives in LDS as bf16; c in registers. Per step: 20 unrolled WMMAs + gates.
// ---------------------------------------------------------------------------
#define RECUR_LDS (102400 + 25600 + 4096)

__global__ __launch_bounds__(160) void k_lstm_recur(
    const float* __restrict__ xg,      // [NT][400], row = n*TT + t
    const __bf16* __restrict__ Wswz,   // swizzled [4][25][32][16] bf16
    float* __restrict__ hout)          // [NT][100]
{
    extern __shared__ char lds[];
    __bf16* sW = (__bf16*)lds;                    // swizzled, 102400 B (offset 0)
    float*  sG = (float*) (lds + 102400);         // [16][400]
    __bf16* sH = (__bf16*)(lds + 102400 + 25600); // [16][128]

    const int tid  = threadIdx.x;
    const int lane = tid & 31;
    const int wave = tid >> 5;          // 0..4
    const int hi = lane >> 4, lo = lane & 15;
    const int nBase = blockIdx.x * 16;

    // ---- TDM: stage swizzled Whh (25600 dwords) into LDS at offset 0.
    if (wave == 0) {
        unsigned long long ga = (unsigned long long)(const void*)Wswz;
        // D# group0: count=1, lds_addr=0, global_addr, type=2
        v4u g0 = { 1u, 0u, (unsigned)ga,
                   (unsigned)((ga >> 32) & 0x01FFFFFFull) | (2u << 30) };
        // D# group1: data_size=4B; tensor_dim0=25600, tensor_dim1=1;
        //            tile_dim0=25600, tile_dim1=1; dim0_stride=25600
        v8i_ g1 = { (int)0x00020000,
                    (int)0x64000000,            // dim0[15:0]=0x6400 in [31:16]
                    (int)0x00010000,            // dim0[31:16]=0 ; dim1 lo16=1
                    (int)0x64000000,            // dim1 hi=0 ; tile_dim0=0x6400
                    (int)0x00000001,            // tile_dim1=1 ; tile_dim2=0
                    (int)25600,                 // dim0_stride lo32
                    0, 0 };
        v4i_ g2 = { 0, 0, 0, 0 };
        v4i_ g3 = { 0, 0, 0, 0 };
#if __clang_major__ >= 23
        v8i_ g4 = { 0, 0, 0, 0, 0, 0, 0, 0 };
        __builtin_amdgcn_tensor_load_to_lds(g0, g1, g2, g3, g4, 0);
#else
        __builtin_amdgcn_tensor_load_to_lds(g0, g1, g2, g3, 0);
#endif
        __builtin_amdgcn_s_wait_tensorcnt(0);
    }
    unsigned* hz = (unsigned*)sH;
    for (int i = tid; i < 1024; i += 160) hz[i] = 0u;
    __syncthreads();

    // per-thread cell state: element idx = tid + 160*r (16*100 = 1600 total)
    float cst[10];
    int erow[10], ecol[10];
    #pragma unroll
    for (int r = 0; r < 10; ++r) {
        cst[r] = 0.0f;
        int idx = tid + 160 * r;
        erow[r] = idx / HH;
        ecol[r] = idx % HH;
    }

    for (int t = 0; t < TT; ++t) {
        // ---- g = h @ WhhT : wave w owns N-tiles {w, w+5, ..., w+20}
        #pragma unroll
        for (int j = 0; j < 5; ++j) {
            const int nt = wave + 5 * j;
            v8f acc = {};
            #pragma unroll
            for (int c = 0; c < 4; ++c) {
                const v8bf* ap = (const v8bf*)(sH + lo * 128 + c * 32 + hi * 16);
                v16bf a = cat16(ap[0], ap[1]);
                const v8bf* bp = (const v8bf*)(sW + (((c * 25) + nt) * 32 + lane) * 16);
                v16bf b = cat16(bp[0], bp[1]);
                acc = __builtin_amdgcn_wmma_f32_16x16x32_bf16(false, a, false, b,
                                                              (short)0, acc, false, false);
            }
            #pragma unroll
            for (int v = 0; v < 8; ++v)
                sG[(v + hi * 8) * G4 + nt * 16 + lo] = acc[v];
        }
        __syncthreads();

        // ---- gates + state update (PyTorch order i,f,g,o)
        #pragma unroll
        for (int r = 0; r < 10; ++r) {
            const int row = erow[r], col = ecol[r];
            const float* xrow = xg + ((size_t)(nBase + row) * TT + t) * G4;
            if (r == 0 && t + 1 < TT)
                __builtin_prefetch(xrow + G4, 0, 0);   // next timestep row
            float gi = sG[row * G4 +        col] + xrow[       col];
            float gf = sG[row * G4 + HH   + col] + xrow[HH   + col];
            float gg = sG[row * G4 + 2*HH + col] + xrow[2*HH + col];
            float go = sG[row * G4 + 3*HH + col] + xrow[3*HH + col];
            float si = 1.0f / (1.0f + __expf(-gi));
            float sf = 1.0f / (1.0f + __expf(-gf));
            float so = 1.0f / (1.0f + __expf(-go));
            float c  = sf * cst[r] + si * tanhf(gg);
            cst[r] = c;
            float h = so * tanhf(c);
            hout[((size_t)(nBase + row) * TT + t) * HH + col] = h;
            sH[row * 128 + col] = (__bf16)h;
        }
        __syncthreads();
    }
}

// ---------------------------------------------------------------------------
// Small FC: out[M][3] = h[M][100] @ W[3][100]^T + b
// ---------------------------------------------------------------------------
__global__ void k_fc3(const float* __restrict__ h, const float* __restrict__ W,
                      const float* __restrict__ b, float* __restrict__ out, int M) {
    int row = blockIdx.x * blockDim.x + threadIdx.x;
    if (row >= M) return;
    const float* hr = h + (size_t)row * HH;
    float s0 = b[0], s1 = b[1], s2 = b[2];
    for (int j = 0; j < HH; ++j) {
        float v = hr[j];
        s0 += v * W[j];
        s1 += v * W[HH + j];
        s2 += v * W[2 * HH + j];
    }
    out[row * 3 + 0] = s0; out[row * 3 + 1] = s1; out[row * 3 + 2] = s2;
}

// ---------------------------------------------------------------------------
// xr = Rz*Ry*Rx * (pts - tr); writes padded bf16 [M][160] directly
// ---------------------------------------------------------------------------
__global__ void k_rotate(const float* __restrict__ x, const float* __restrict__ rot,
                         const float* __restrict__ tr, __bf16* __restrict__ xrbf, int M) {
    int r = blockIdx.x * blockDim.x + threadIdx.x;
    if (r >= M) return;
    float a = rot[r*3+0], b = rot[r*3+1], c = rot[r*3+2];
    float ca = __cosf(a), sa = __sinf(a);
    float cb = __cosf(b), sb = __sinf(b);
    float cc = __cosf(c), sc = __sinf(c);
    float t0 = tr[r*3+0], t1 = tr[r*3+1], t2 = tr[r*3+2];
    const float* xp = x + (size_t)r * SS;
    __bf16* xo = xrbf + (size_t)r * 160;
    for (int p = 0; p < 50; ++p) {
        float p0 = xp[p*3+0] - t0, p1 = xp[p*3+1] - t1, p2 = xp[p*3+2] - t2;
        float q1 =  ca*p1 + sa*p2, q2 = -sa*p1 + ca*p2;          // Rx
        float r0 =  cb*p0 - sb*q2, r2 =  sb*p0 + cb*q2;          // Ry
        xo[p*3+0] = (__bf16)( cc*r0 + sc*q1);                    // Rz
        xo[p*3+1] = (__bf16)(-sc*r0 + cc*q1);
        xo[p*3+2] = (__bf16)( r2);
    }
    #pragma unroll
    for (int k = SS; k < 160; ++k) xo[k] = (__bf16)0.0f;
}

// ---------------------------------------------------------------------------
__global__ void k_mean(const float* __restrict__ h, float* __restrict__ hbar) {
    int i = blockIdx.x * blockDim.x + threadIdx.x;
    if (i >= NB * HH) return;
    int n = i / HH, j = i % HH;
    const float* p = h + (size_t)n * TT * HH + j;
    float s = 0.0f;
    for (int t = 0; t < TT; ++t) s += p[t * HH];
    hbar[i] = s * (1.0f / TT);
}

__global__ void k_fc_softmax(const float* __restrict__ hbar, const float* __restrict__ W,
                             const float* __restrict__ b, float* __restrict__ out) {
    __shared__ float logit[64];
    __shared__ float red[2];
    int n = blockIdx.x, c = threadIdx.x;
    float v = -1e30f;
    if (c < NCLS) {
        float s = b[c];
        const float* hr = hbar + n * HH;
        const float* wr = W + c * HH;
        for (int j = 0; j < HH; ++j) s += hr[j] * wr[j];
        v = s;
    }
    logit[c] = v;
    __syncthreads();
    if (c == 0) {
        float m = -1e30f;
        for (int j = 0; j < NCLS; ++j) m = fmaxf(m, logit[j]);
        float sum = 0.0f;
        for (int j = 0; j < NCLS; ++j) sum += __expf(logit[j] - m);
        red[0] = m; red[1] = sum;
    }
    __syncthreads();
    if (c < NCLS) out[n * NCLS + c] = __expf(logit[c] - red[0]) / red[1];
}

// ---------------------------------------------------------------------------
extern "C" void kernel_launch(void* const* d_in, const int* in_sizes, int n_in,
                              void* d_out, int out_size, void* d_ws, size_t ws_size,
                              hipStream_t stream) {
    (void)in_sizes; (void)n_in; (void)out_size; (void)ws_size;

    hipFuncSetAttribute(reinterpret_cast<const void*>(k_lstm_recur),
                        hipFuncAttributeMaxDynamicSharedMemorySize, RECUR_LDS);

    const float* x = (const float*)d_in[0];
    struct LP { const float *Wih, *Whh, *bih, *bhh; int K, KC32; };
    LP lp[5] = {
        { (const float*)d_in[1],  (const float*)d_in[2],  (const float*)d_in[3],  (const float*)d_in[4],  SS, 5 }, // rot
        { (const float*)d_in[5],  (const float*)d_in[6],  (const float*)d_in[7],  (const float*)d_in[8],  SS, 5 }, // tr
        { (const float*)d_in[13], (const float*)d_in[14], (const float*)d_in[15], (const float*)d_in[16], SS, 5 }, // main0
        { (const float*)d_in[17], (const float*)d_in[18], (const float*)d_in[19], (const float*)d_in[20], HH, 4 }, // main1
        { (const float*)d_in[21], (const float*)d_in[22], (const float*)d_in[23], (const float*)d_in[24], HH, 4 }, // main2
    };
    const float* rot_fcW = (const float*)d_in[9];
    const float* rot_fcb = (const float*)d_in[10];
    const float* tr_fcW  = (const float*)d_in[11];
    const float* tr_fcb  = (const float*)d_in[12];
    const float* m_fcW   = (const float*)d_in[25];
    const float* m_fcb   = (const float*)d_in[26];

    // ---- workspace carve-up
    char* ws = (char*)d_ws;
    size_t cur = 0;
    auto alloc = [&](size_t bytes) -> char* {
        char* p = ws + cur;
        cur = (cur + bytes + 255) & ~(size_t)255;
        return p;
    };
    float*  xg   = (float*)alloc((size_t)NT * G4 * 4);
    float*  hA   = (float*)alloc((size_t)NT * HH * 4);
    float*  hB   = (float*)alloc((size_t)NT * HH * 4);
    float*  rot3 = (float*)alloc((size_t)NT * 3 * 4);
    float*  tr3  = (float*)alloc((size_t)NT * 3 * 4);
    float*  hbar = (float*)alloc((size_t)NB * HH * 4);
    __bf16* xbf  = (__bf16*)alloc((size_t)NT * 160 * 2);   // x / xr (reused)
    __bf16* hbf  = (__bf16*)alloc((size_t)NT * 128 * 2);   // hA / hB (reused)
    __bf16* wtih[5]; __bf16* wthh[5]; float* bcomb[5];
    for (int l = 0; l < 5; ++l) {
        wtih[l]  = (__bf16*)alloc((size_t)lp[l].KC32 * 25 * 512 * 2);
        wthh[l]  = (__bf16*)alloc((size_t)4 * 25 * 512 * 2);
        bcomb[l] = (float*)alloc(G4 * 4);
    }

    // ---- weight conversion (fragment-swizzled bf16)
    for (int l = 0; l < 5; ++l) {
        int tot = lp[l].KC32 * 25 * 512;
        k_convert_w_swz<<<(tot + 255) / 256, 256, 0, stream>>>(lp[l].Wih, wtih[l], lp[l].K, lp[l].KC32);
        k_convert_w_swz<<<(4 * 25 * 512 + 255) / 256, 256, 0, stream>>>(lp[l].Whh, wthh[l], HH, 4);
        k_bias_add<<<2, 256, 0, stream>>>(lp[l].bih, lp[l].bhh, bcomb[l], G4);
    }

    dim3 gGemm(NT / 16, G4 / 16);
    auto gemm160 = [&](const __bf16* A, int l) {
        k_gemm_bf16<5><<<gGemm, 32, 0, stream>>>(A, wtih[l], bcomb[l], xg);
    };
    auto gemm128 = [&](const __bf16* A, int l) {
        k_gemm_bf16<4><<<gGemm, 32, 0, stream>>>(A, wtih[l], bcomb[l], xg);
    };
    auto recur = [&](int l, float* hout) {
        k_lstm_recur<<<NB / 16, 160, RECUR_LDS, stream>>>(xg, wthh[l], hout);
    };
    auto cvtact = [&](const float* src, __bf16* dst, int K, int Kpad) {
        int tot = NT * Kpad;
        k_convert_act<<<(tot + 255) / 256, 256, 0, stream>>>(src, dst, NT, K, Kpad);
    };

    // ---- rot LSTM -> angles ; tr LSTM -> translations
    cvtact(x, xbf, SS, 160);
    gemm160(xbf, 0); recur(0, hA);
    k_fc3<<<NT / 256, 256, 0, stream>>>(hA, rot_fcW, rot_fcb, rot3, NT);
    gemm160(xbf, 1); recur(1, hA);
    k_fc3<<<NT / 256, 256, 0, stream>>>(hA, tr_fcW, tr_fcb, tr3, NT);
    // ---- rotate points (writes padded bf16 directly)
    k_rotate<<<NT / 256, 256, 0, stream>>>(x, rot3, tr3, xbf, NT);
    // ---- main 3-layer LSTM
    gemm160(xbf, 2); recur(2, hA);
    cvtact(hA, hbf, HH, 128);
    gemm128(hbf, 3); recur(3, hB);
    cvtact(hB, hbf, HH, 128);
    gemm128(hbf, 4); recur(4, hA);
    // ---- head
    k_mean<<<(NB * HH + 255) / 256, 256, 0, stream>>>(hA, hbar);
    k_fc_softmax<<<NB, 64, 0, stream>>>(hbar, m_fcW, m_fcb, (float*)d_out);
}
